// Multihead_self_attention_16174846836968
// MI455X (gfx1250) — matrix operand order repaired and checked by reading, not compiled, not run on hardware
//
#include <hip/hip_runtime.h>

typedef __attribute__((ext_vector_type(16))) _Float16 v16h;
typedef __attribute__((ext_vector_type(8)))  float    v8f;

#define D_MODEL 1024
#define NHEAD   16
#define DK      64
#define BATCH   2
#define SEQ     2048
#define MROWS   (BATCH * SEQ)   // 4096

union Frag16 { v16h v; uint4 q[2]; };

// ---------------------------------------------------------------------------
// f32 -> f16 convert (same layout)
// ---------------------------------------------------------------------------
__global__ __launch_bounds__(256) void cvt_f16_kernel(const float* __restrict__ src,
                                                      _Float16* __restrict__ dst, int n) {
    int i = blockIdx.x * blockDim.x + threadIdx.x;
    if (i < n) dst[i] = (_Float16)src[i];
}

// ---------------------------------------------------------------------------
// W[k][n] f32 -> WT[n][k] f16 (so WMMA B-fragments are contiguous per lane)
// ---------------------------------------------------------------------------
__global__ __launch_bounds__(256) void transpose_w_kernel(const float* __restrict__ W,
                                                          _Float16* __restrict__ WT) {
    int i = blockIdx.x * blockDim.x + threadIdx.x;   // i over D*D
    int k = i >> 10;
    int n = i & (D_MODEL - 1);
    WT[(size_t)n * D_MODEL + k] = (_Float16)W[i];
}

// ---------------------------------------------------------------------------
// C[M,N] f32 = A[M,K] f16 (row-major) x BT[N,K] f16 (row-major, pre-transposed)
// Register-blocked: one wave computes a 32x64 output region (2 m-tiles x
// 4 n-tiles). Per 32-wide k-step: 2 A-frags + 4 B-frags (6 KB) -> 8 WMMAs
// (128 KFLOP), ~21 FLOP/byte of fragment traffic.
// grid = (M/32, N/(64*8)), block = 256 (8 waves span 512 columns; the two
// A row-tiles are shared by all waves of the block through WGP$/L2).
// ---------------------------------------------------------------------------
__global__ __launch_bounds__(256) void gemm_f16_kernel(const _Float16* __restrict__ A,
                                                       const _Float16* __restrict__ BT,
                                                       float* __restrict__ C) {
    const int lane = threadIdx.x & 31;
    const int wave = threadIdx.x >> 5;
    const int half = lane >> 4;
    const int l16  = lane & 15;
    const int m0 = blockIdx.x * 32;                 // two 16-row tiles
    const int n0 = (blockIdx.y * 8 + wave) * 64;    // four 16-col tiles

    const _Float16* arow0 = A + (size_t)(m0 + l16) * D_MODEL;
    const _Float16* arow1 = A + (size_t)(m0 + 16 + l16) * D_MODEL;
    const _Float16* brow  = BT + (size_t)(n0 + l16) * D_MODEL;

    v8f acc[2][4] = {};
    for (int k = 0; k < D_MODEL; k += 32) {
        Frag16 a0, a1, b[4];
        // A layout: lane holds row m=l16; halves = two contiguous 8-runs of k
        const _Float16* ap0 = arow0 + k + half * 8;
        a0.q[0] = *(const uint4*)ap0;
        a0.q[1] = *(const uint4*)(ap0 + 16);
        const _Float16* ap1 = arow1 + k + half * 8;
        a1.q[0] = *(const uint4*)ap1;
        a1.q[1] = *(const uint4*)(ap1 + 16);
        // B layout: lane holds col n=l16; halves = two contiguous 8-runs of k
#pragma unroll
        for (int j = 0; j < 4; ++j) {
            const _Float16* bp = brow + (size_t)(j * 16) * D_MODEL + k + half * 8;
            b[j].q[0] = *(const uint4*)bp;
            b[j].q[1] = *(const uint4*)(bp + 16);
        }
#pragma unroll
        for (int j = 0; j < 4; ++j) {
            acc[0][j] = __builtin_amdgcn_wmma_f32_16x16x32_f16(false, a0.v, false, b[j].v,
                                                               (short)0, acc[0][j], false, false);
            acc[1][j] = __builtin_amdgcn_wmma_f32_16x16x32_f16(false, a1.v, false, b[j].v,
                                                               (short)0, acc[1][j], false, false);
        }
    }
    // C/D layout: vgpr r -> m = r + half*8 ; lane -> n = l16
#pragma unroll
    for (int i = 0; i < 2; ++i) {
        float* cp = C + (size_t)(m0 + i * 16 + half * 8) * D_MODEL + n0 + l16;
#pragma unroll
        for (int r = 0; r < 8; ++r)
#pragma unroll
            for (int j = 0; j < 4; ++j)
                cp[(size_t)r * D_MODEL + j * 16] = acc[i][j][r];
    }
}

// ---------------------------------------------------------------------------
// RoPE on Q,K (f32 in) -> f16 out (1/sqrt(dk) folded into Q);
// V converted to f16 transposed per head: Vt[b][h][d][s].
// One thread per (b, s, h, i) with i in [0, dk/2).
// ---------------------------------------------------------------------------
__global__ __launch_bounds__(256) void rope_cvt_kernel(const float* __restrict__ Qf,
                                                       const float* __restrict__ Kf,
                                                       const float* __restrict__ Vf,
                                                       const int* __restrict__ pos,
                                                       _Float16* __restrict__ Qh,
                                                       _Float16* __restrict__ Kh,
                                                       _Float16* __restrict__ Vt) {
    int idx = blockIdx.x * blockDim.x + threadIdx.x;   // B*S*H*32 threads
    int i = idx & 31;            // pair index within head
    int h = (idx >> 5) & 15;
    int s = (idx >> 9) & (SEQ - 1);
    int b = idx >> 20;

    float p    = (float)pos[b * SEQ + s];
    float freq = __powf(10000.0f, -(float)(2 * i) * (1.0f / 64.0f));
    float sn, cs;
    __sincosf(p * freq, &sn, &cs);

    size_t base = (size_t)(b * SEQ + s) * D_MODEL + h * DK + 2 * i;
    float q1 = Qf[base], q2 = Qf[base + 1];
    Qh[base]     = (_Float16)((q1 * cs - q2 * sn) * 0.125f);  // 1/sqrt(64)
    Qh[base + 1] = (_Float16)((q1 * sn + q2 * cs) * 0.125f);
    float k1 = Kf[base], k2 = Kf[base + 1];
    Kh[base]     = (_Float16)(k1 * cs - k2 * sn);
    Kh[base + 1] = (_Float16)(k1 * sn + k2 * cs);

    size_t vb = ((size_t)((b * NHEAD + h) * DK) + 2 * i) * SEQ + s;
    Vt[vb]       = (_Float16)Vf[base];
    Vt[vb + SEQ] = (_Float16)Vf[base + 1];
}

// ---------------------------------------------------------------------------
// Causal flash attention. One wave per (16-query tile, head, batch).
// Per 32-key block: 4 QK^T WMMAs (2 subtiles x 2 k-steps of dk), online
// softmax (shfl_xor row reductions), P relayout D->A via LDS (+s_wait_dscnt),
// 4 PV WMMAs. Output written as f16 [B,S,D] for the Wo GEMM.
// grid = (SEQ/16/8, NHEAD, BATCH), block = 256.
// ---------------------------------------------------------------------------
__global__ __launch_bounds__(256) void attn_kernel(const _Float16* __restrict__ Qh,
                                                   const _Float16* __restrict__ Kh,
                                                   const _Float16* __restrict__ Vt,
                                                   _Float16* __restrict__ atth) {
    __shared__ __align__(16) _Float16 plds[8][16 * 32];   // per-wave 16x32 P tile

    const int lane = threadIdx.x & 31;
    const int wave = threadIdx.x >> 5;
    const int half = lane >> 4;
    const int l16  = lane & 15;
    const int qtile = blockIdx.x * 8 + wave;
    const int h = blockIdx.y;
    const int b = blockIdx.z;
    const int q0 = qtile * 16;
    const float NEG_INF = -__builtin_inff();

    // Q A-fragments (16 queries x dk=64 -> two k-steps of 32)
    Frag16 qa[2];
    const _Float16* qrow = Qh + (size_t)(b * SEQ + q0 + l16) * D_MODEL + h * DK;
#pragma unroll
    for (int s = 0; s < 2; ++s) {
        const _Float16* ap = qrow + s * 32 + half * 8;
        qa[s].q[0] = *(const uint4*)ap;
        qa[s].q[1] = *(const uint4*)(ap + 16);
    }

    v8f oacc[4] = {};          // 16 queries x 64 d, 4 chunks of 16 cols
    float rmax[8], rsum[8];
#pragma unroll
    for (int r = 0; r < 8; ++r) { rmax[r] = NEG_INF; rsum[r] = 0.0f; }

    const _Float16* kbase = Kh + (size_t)(b * SEQ) * D_MODEL + h * DK;
    const _Float16* vbase = Vt + (size_t)((b * NHEAD + h) * DK) * SEQ;

    for (int kb = 0; kb < q0 + 16; kb += 32) {
        v8f c0 = {}, c1 = {};
        // scores: keys [kb, kb+16)
#pragma unroll
        for (int s = 0; s < 2; ++s) {
            Frag16 kf;
            const _Float16* kp = kbase + (size_t)(kb + l16) * D_MODEL + s * 32 + half * 8;
            kf.q[0] = *(const uint4*)kp;
            kf.q[1] = *(const uint4*)(kp + 16);
            c0 = __builtin_amdgcn_wmma_f32_16x16x32_f16(false, qa[s].v, false, kf.v,
                                                        (short)0, c0, false, false);
        }
        // scores: keys [kb+16, kb+32) — only if any of them is unmasked
        if (kb < q0) {
#pragma unroll
            for (int s = 0; s < 2; ++s) {
                Frag16 kf;
                const _Float16* kp = kbase + (size_t)(kb + 16 + l16) * D_MODEL + s * 32 + half * 8;
                kf.q[0] = *(const uint4*)kp;
                kf.q[1] = *(const uint4*)(kp + 16);
                c1 = __builtin_amdgcn_wmma_f32_16x16x32_f16(false, qa[s].v, false, kf.v,
                                                            (short)0, c1, false, false);
            }
        }
        // online softmax; D layout: vgpr r -> row m = r + half*8, lane -> key n = l16
#pragma unroll
        for (int r = 0; r < 8; ++r) {
            const int m = r + half * 8;
            const int q = q0 + m;
            if (kb + l16 > q)      c0[r] = NEG_INF;
            if (kb + 16 + l16 > q) c1[r] = NEG_INF;
            float mx = fmaxf(c0[r], c1[r]);
            mx = fmaxf(mx, __shfl_xor(mx, 1, 32));
            mx = fmaxf(mx, __shfl_xor(mx, 2, 32));
            mx = fmaxf(mx, __shfl_xor(mx, 4, 32));
            mx = fmaxf(mx, __shfl_xor(mx, 8, 32));
            const float nmax = fmaxf(rmax[r], mx);
            const float corr = __expf(rmax[r] - nmax);
            const float p0 = __expf(c0[r] - nmax);
            const float p1 = __expf(c1[r] - nmax);
            float sm = p0 + p1;
            sm += __shfl_xor(sm, 1, 32);
            sm += __shfl_xor(sm, 2, 32);
            sm += __shfl_xor(sm, 4, 32);
            sm += __shfl_xor(sm, 8, 32);
            rsum[r] = rsum[r] * corr + sm;
            rmax[r] = nmax;
            oacc[0][r] *= corr; oacc[1][r] *= corr;
            oacc[2][r] *= corr; oacc[3][r] *= corr;
            plds[wave][m * 32 + l16]      = (_Float16)p0;
            plds[wave][m * 32 + 16 + l16] = (_Float16)p1;
        }
        // wave-local LDS RAW: drain DS counter before re-reading P in A layout
        asm volatile("s_wait_dscnt 0" ::: "memory");
        Frag16 pa;
        const _Float16* pp = &plds[wave][l16 * 32 + half * 8];
        pa.q[0] = *(const uint4*)pp;
        pa.q[1] = *(const uint4*)(pp + 16);
        // O += P (16x32) x V (32 x 64): 4 WMMAs over d-chunks of 16
#pragma unroll
        for (int c = 0; c < 4; ++c) {
            Frag16 vf;
            const _Float16* vp = vbase + (size_t)(c * 16 + l16) * SEQ + kb + half * 8;
            vf.q[0] = *(const uint4*)vp;
            vf.q[1] = *(const uint4*)(vp + 16);
            oacc[c] = __builtin_amdgcn_wmma_f32_16x16x32_f16(false, pa.v, false, vf.v,
                                                             (short)0, oacc[c], false, false);
        }
    }

    // normalize and store f16 att [B,S,D]
    _Float16* orow = atth + (size_t)(b * SEQ + q0 + half * 8) * D_MODEL + h * DK + l16;
#pragma unroll
    for (int r = 0; r < 8; ++r) {
        const float inv = 1.0f / rsum[r];
#pragma unroll
        for (int c = 0; c < 4; ++c)
            orow[(size_t)r * D_MODEL + c * 16] = (_Float16)(oacc[c][r] * inv);
    }
}

// ---------------------------------------------------------------------------
// host-side orchestration
// ---------------------------------------------------------------------------
extern "C" void kernel_launch(void* const* d_in, const int* in_sizes, int n_in,
                              void* d_out, int out_size, void* d_ws, size_t ws_size,
                              hipStream_t stream) {
    (void)in_sizes; (void)n_in; (void)out_size; (void)ws_size;
    const float* x  = (const float*)d_in[0];
    const float* Wq = (const float*)d_in[1];
    const float* Wk = (const float*)d_in[2];
    const float* Wv = (const float*)d_in[3];
    const float* Wo = (const float*)d_in[4];
    const int*  pos = (const int*)d_in[5];
    float* out = (float*)d_out;

    char* w = (char*)d_ws;
    const size_t MB = 1ull << 20;
    _Float16* xh   = (_Float16*)(w + 0 * MB);    // 8 MiB
    _Float16* wqT  = (_Float16*)(w + 8 * MB);    // 2 MiB each
    _Float16* wkT  = (_Float16*)(w + 10 * MB);
    _Float16* wvT  = (_Float16*)(w + 12 * MB);
    _Float16* woT  = (_Float16*)(w + 14 * MB);
    float*    qf   = (float*)(w + 16 * MB);      // 16 MiB each
    float*    kf   = (float*)(w + 32 * MB);
    float*    vf   = (float*)(w + 48 * MB);
    _Float16* qh   = (_Float16*)(w + 64 * MB);   // 8 MiB each
    _Float16* kh   = (_Float16*)(w + 72 * MB);
    _Float16* vt   = (_Float16*)(w + 80 * MB);
    _Float16* atth = (_Float16*)(w + 88 * MB);   // total 96 MiB

    // 1) precision conversion + weight transposes
    cvt_f16_kernel<<<(MROWS * D_MODEL) / 256, 256, 0, stream>>>(x, xh, MROWS * D_MODEL);
    transpose_w_kernel<<<(D_MODEL * D_MODEL) / 256, 256, 0, stream>>>(Wq, wqT);
    transpose_w_kernel<<<(D_MODEL * D_MODEL) / 256, 256, 0, stream>>>(Wk, wkT);
    transpose_w_kernel<<<(D_MODEL * D_MODEL) / 256, 256, 0, stream>>>(Wv, wvT);
    transpose_w_kernel<<<(D_MODEL * D_MODEL) / 256, 256, 0, stream>>>(Wo, woT);

    // 2) QKV projections (register-blocked WMMA GEMMs, 32x64 per wave)
    dim3 gg(MROWS / 32, D_MODEL / (64 * 8));
    gemm_f16_kernel<<<gg, 256, 0, stream>>>(xh, wqT, qf);
    gemm_f16_kernel<<<gg, 256, 0, stream>>>(xh, wkT, kf);
    gemm_f16_kernel<<<gg, 256, 0, stream>>>(xh, wvT, vf);

    // 3) RoPE + f16 conversion (+ V head-transpose)
    rope_cvt_kernel<<<(BATCH * SEQ * NHEAD * 32) / 256, 256, 0, stream>>>(
        qf, kf, vf, pos, qh, kh, vt);

    // 4) causal flash attention (WMMA for QK^T and PV)
    attn_kernel<<<dim3(SEQ / (16 * 8), NHEAD, BATCH), 256, 0, stream>>>(qh, kh, vt, atth);

    // 5) output projection
    gemm_f16_kernel<<<gg, 256, 0, stream>>>(atth, woT, out);
}
